// DAGController_15255723835585
// MI455X (gfx1250) — matrix operand order repaired and checked by reading, not compiled, not run on hardware
//
#include <hip/hip_runtime.h>

#define NSPANS 48
#define HIDN   64
#define NDEC   1128
#define BATCHN 2048
#define EPSV   1e-8f

typedef __attribute__((ext_vector_type(16))) _Float16 v16h;
typedef __attribute__((ext_vector_type(8)))  _Float16 v8h;
typedef __attribute__((ext_vector_type(8)))  float    v8f;
typedef __attribute__((ext_vector_type(2)))  float    v2f;

__device__ __forceinline__ float sigmoid_f(float x) {
    return __builtin_amdgcn_rcpf(1.0f + __expf(-x));
}
__device__ __forceinline__ float tanh_f(float x) {
    // tanh(x) = 2*sigmoid(2x) - 1
    return 2.0f * __builtin_amdgcn_rcpf(1.0f + __expf(-2.0f * x)) - 1.0f;
}

// One wave (32 lanes) owns 16 batch rows for the whole 1128-step scan.
// GEMM is done transposed: gh^T = w_hh(192x64) @ h^T(64x16):
//   A = w_hh tiles (f16, register resident), B = h^T (rebuilt from LDS each step),
//   C layout: element (j = tile*16 + v + 8*(lane/16), batch n = lane&15).
__global__ __launch_bounds__(32)
void gru_dag_kernel(const float* __restrict__ embed,
                    const float* __restrict__ w_ih,
                    const float* __restrict__ w_hh,
                    const float* __restrict__ b_ih,
                    const float* __restrict__ b_hh,
                    const float* __restrict__ w_head,
                    const float* __restrict__ b_head,
                    const float* __restrict__ u,
                    int*   __restrict__ out_dec,
                    float* __restrict__ out_lp)
{
    __shared__ float    e0s[NSPANS];
    __shared__ _Float16 hbuf[16 * 72];   // h^T as [n][k], stride 72 halfs (bank-spread)

    const int lane = threadIdx.x & 31;
    const int nn   = lane & 15;          // batch column within tile
    const int hf   = lane >> 4;          // lane half
    const int brow = blockIdx.x * 16 + nn;

    for (int s = lane; s < NSPANS; s += 32)  e0s[s]  = embed[2 * s];
    for (int s = lane; s < 16 * 72; s += 32) hbuf[s] = (_Float16)0.0f;

    // ---- A fragments: w_hh (192x64) as f16 16x16x32 A tiles (loop invariant) ----
    // A 16-bit layout: lane row m = lane&15; lanes0-15: V0..3=k0..7, V4..7=k16..23;
    // lanes16-31: V0..3=k8..15, V4..7=k24..31 (plus frag base f*32).
    v16h Ah[12][2];
#pragma unroll
    for (int t = 0; t < 12; ++t) {
        const int j = t * 16 + nn;
#pragma unroll
        for (int f = 0; f < 2; ++f) {
#pragma unroll
            for (int idx = 0; idx < 16; ++idx) {
                int k = f * 32 + ((idx < 8) ? (hf * 8 + idx)
                                            : (16 + hf * 8 + (idx - 8)));
                Ah[t][f][idx] = (_Float16)w_hh[j * HIDN + k];
            }
        }
    }

    // ---- A fragments for the input/bias term: f32 16x16x4 ----
    // Columns of the virtual K=4 input: [w_ih[:,0](sv), w_ih[:,1](dv), w_ih[:,2](prev), bias].
    // f32 A 16x4 layout: lanes0-15: V0=K0,V1=K1; lanes16-31: V0=K2,V1=K3.
    v2f Ax[12];
#pragma unroll
    for (int t = 0; t < 12; ++t) {
        const int j = t * 16 + nn;
        float w0 = w_ih[j * 3 + 0];
        float w1 = w_ih[j * 3 + 1];
        float w2 = w_ih[j * 3 + 2];
        // r/z gates: full bias folded; n gate: only b_ih (b_hh must stay outside,
        // because r multiplies (h-part + b_hh) before adding the input part).
        float bias = (t < 8) ? (b_ih[j] + b_hh[j]) : b_ih[j];
        Ax[t][0] = (hf == 0) ? w0 : w2;
        Ax[t][1] = (hf == 0) ? w1 : bias;
    }

    // Per-lane constants in C layout (j = base + 8*hf + v)
    float bhh_n[4][8], whd[4][8];
#pragma unroll
    for (int t = 0; t < 4; ++t)
#pragma unroll
        for (int v = 0; v < 8; ++v) {
            bhh_n[t][v] = b_hh[128 + t * 16 + 8 * hf + v];
            whd[t][v]   = w_head[t * 16 + 8 * hf + v];
        }
    const float bh = b_head[0];

    v8f hc[4];                           // current h in C layout (f32)
#pragma unroll
    for (int t = 0; t < 4; ++t) { v8f z = {}; hc[t] = z; }
    float prev = 0.0f;

    int tix = 0;
    for (int i = 0; i < NSPANS; ++i) {
        const float sv = e0s[i];
        for (int jj = i + 1; jj < NSPANS; ++jj) {
            const float dv = e0s[jj];

            // ---- B fragments ----
            // x-part (f32 4x16): V0 = k0(sv) | k2(prev[n]); V1 = k1(dv) | k3(1.0)
            v2f bx;
            bx[0] = (hf == 0) ? sv : prev;
            bx[1] = (hf == 0) ? dv : 1.0f;

            // h^T (f16 32x16 x2): lanes0-15 k0..7,k16..23 / lanes16-31 k8..15,k24..31 (frag0), +32 (frag1)
            v16h B0, B1;
            {
                const int base = nn * 72 + hf * 8;
                v8h l0 = *(const v8h*)&hbuf[base];
                v8h l1 = *(const v8h*)&hbuf[base + 16];
                v8h l2 = *(const v8h*)&hbuf[base + 32];
                v8h l3 = *(const v8h*)&hbuf[base + 48];
#pragma unroll
                for (int k = 0; k < 8; ++k) {
                    B0[k] = l0[k]; B0[k + 8] = l1[k];
                    B1[k] = l2[k]; B1[k + 8] = l3[k];
                }
            }

            // ---- r, z gates: tiles 0..7, full preactivation in one WMMA chain ----
            v8f rt[4], zt[4];
#pragma unroll
            for (int t = 0; t < 8; ++t) {
                v8f c = {};
                c = __builtin_amdgcn_wmma_f32_16x16x4_f32(false, Ax[t], false, bx,
                                                          (short)0, c, false, false);
                c = __builtin_amdgcn_wmma_f32_16x16x32_f16(false, Ah[t][0], false, B0,
                                                           (short)0, c, false, false);
                c = __builtin_amdgcn_wmma_f32_16x16x32_f16(false, Ah[t][1], false, B1,
                                                           (short)0, c, false, false);
#pragma unroll
                for (int v = 0; v < 8; ++v) {
                    float g = sigmoid_f(c[v]);
                    if (t < 4) rt[t][v] = g; else zt[t - 4][v] = g;
                }
            }

            // ---- n gate + h update: tiles 8..11 (input and hidden parts separate) ----
#pragma unroll
            for (int t = 0; t < 4; ++t) {
                const int tt = 8 + t;
                v8f ci = {};
                ci = __builtin_amdgcn_wmma_f32_16x16x4_f32(false, Ax[tt], false, bx,
                                                           (short)0, ci, false, false);
                v8f ch = {};
                ch = __builtin_amdgcn_wmma_f32_16x16x32_f16(false, Ah[tt][0], false, B0,
                                                            (short)0, ch, false, false);
                ch = __builtin_amdgcn_wmma_f32_16x16x32_f16(false, Ah[tt][1], false, B1,
                                                            (short)0, ch, false, false);
                v8h ph;
#pragma unroll
                for (int v = 0; v < 8; ++v) {
                    float npre = ci[v] + rt[t][v] * (ch[v] + bhh_n[t][v]);
                    float nv = tanh_f(npre);
                    float z  = zt[t][v];
                    float h  = (1.0f - z) * nv + z * hc[t][v];
                    hc[t][v] = h;
                    ph[v] = (_Float16)h;
                }
                // store h_new (f16) for next step's B fragments; packed 8 halfs = b128
                *(v8h*)&hbuf[nn * 72 + t * 16 + 8 * hf] = ph;
            }

            // ---- head: logit[n] = sum_j h[j][n]*w_head[j] + bh ----
            float part = 0.0f;
#pragma unroll
            for (int t = 0; t < 4; ++t)
#pragma unroll
                for (int v = 0; v < 8; ++v)
                    part += hc[t][v] * whd[t][v];
            float logit = part + __shfl_xor(part, 16, 32) + bh;

            float prob = sigmoid_f(logit);
            float uu   = u[(size_t)brow * NDEC + tix];
            float dec  = (uu < prob) ? 1.0f : 0.0f;
            float psel = (dec != 0.0f) ? (prob + EPSV) : (1.0f - prob + EPSV);
            float lp   = __logf(psel);

            if (hf == 0) {
                out_dec[(size_t)brow * NDEC + tix] = (int)dec;
                out_lp [(size_t)brow * NDEC + tix] = lp;
            }
            prev = dec;
            ++tix;
        }
    }
}

extern "C" void kernel_launch(void* const* d_in, const int* in_sizes, int n_in,
                              void* d_out, int out_size, void* d_ws, size_t ws_size,
                              hipStream_t stream) {
    const float* embed  = (const float*)d_in[0];
    const float* w_ih   = (const float*)d_in[1];
    const float* w_hh   = (const float*)d_in[2];
    const float* b_ih   = (const float*)d_in[3];
    const float* b_hh   = (const float*)d_in[4];
    const float* w_head = (const float*)d_in[5];
    const float* b_head = (const float*)d_in[6];
    const float* u      = (const float*)d_in[7];

    int*   out_dec = (int*)d_out;                                   // decs (B,NDEC) int32
    float* out_lp  = (float*)d_out + (size_t)BATCHN * NDEC;         // lps  (B,NDEC) f32

    dim3 grid(BATCHN / 16);
    gru_dag_kernel<<<grid, 32, 0, stream>>>(embed, w_ih, w_hh, b_ih, b_hh,
                                            w_head, b_head, u, out_dec, out_lp);
}
